// Level1GNN_46042049413292
// MI455X (gfx1250) — matrix operand order repaired and checked by reading, not compiled, not run on hardware
//
#include <hip/hip_runtime.h>
#include <hip/hip_bf16.h>
#include <math.h>

#define N_NODES 50000
#define N_EDGES 800000
#define IN_DIM  64
#define HID     128
#define N_LAYERS 3
#define N_GRAPHS 64

typedef __bf16 bf16_t;
typedef __attribute__((ext_vector_type(16))) __bf16 v16bf;
typedef __attribute__((ext_vector_type(8)))  float  v8f;

// ---------------------------------------------------------------- utilities
__global__ void k_zero(float* __restrict__ p, unsigned n) {
    unsigned i = blockIdx.x * 256u + threadIdx.x;
    if (i < n) p[i] = 0.0f;
}

// ---------------------------------------------------------------- input projection: h = x @ Wp + bp
// grid = 3125 blocks (one 16-row M tile each), 256 threads = 8 waves = 8 N tiles
__global__ void __launch_bounds__(256) k_proj(const float* __restrict__ x,
                                              const float* __restrict__ Wp,
                                              const float* __restrict__ bp,
                                              float* __restrict__ h) {
    __shared__ bf16_t sA[16 * 64];     // x tile, row-major [16][64]
    __shared__ bf16_t sBT[128 * 64];   // Wp transposed [n][k]
    const unsigned tid = threadIdx.x;
    const unsigned mbase = blockIdx.x * 16u;

    for (unsigned idx = tid; idx < 64u * 128u; idx += 256u) {
        unsigned k = idx >> 7, n = idx & 127u;
        sBT[n * 64u + k] = (bf16_t)Wp[idx];
    }
    for (unsigned idx = tid; idx < 16u * 64u; idx += 256u) {
        unsigned r = idx >> 6, k = idx & 63u;
        sA[idx] = (bf16_t)x[(size_t)(mbase + r) * 64u + k];
    }
    __syncthreads();

    const unsigned wave = tid >> 5, lane = tid & 31u;
    const unsigned half = lane >> 4, r = lane & 15u;   // A row / D column-in-tile
    const unsigned n = wave * 16u + r;                 // output column

    v8f acc = {};
#pragma unroll
    for (unsigned kt = 0; kt < 2; ++kt) {
        const unsigned kb = kt * 32u;
        v16bf a, b;
#pragma unroll
        for (int e = 0; e < 8; ++e) {
            a[e]     = sA[r * 64u + kb + half * 8u + e];        // K = kb + half*8 + e
            a[e + 8] = sA[r * 64u + kb + 16u + half * 8u + e];  // K = kb + 16 + half*8 + e
        }
#pragma unroll
        for (int e = 0; e < 16; ++e)
            b[e] = sBT[n * 64u + kb + half * 8u + (e & 7) + (e >> 3) * 16u];  // K = kb + (e>>3)*16 + half*8 + (e&7)
        acc = __builtin_amdgcn_wmma_f32_16x16x32_bf16(false, a, false, b,
                                                      (short)0, acc, false, false);
    }
    const float bias = bp[n];
#pragma unroll
    for (int v = 0; v < 8; ++v) {
        unsigned row = mbase + half * 8u + (unsigned)v;
        h[(size_t)row * 128u + n] = acc[v] + bias;
    }
}

// ---------------------------------------------------------------- degree / edge scatter
__global__ void k_deg(const int* __restrict__ ei, float* __restrict__ deg) {
    unsigned e = blockIdx.x * 256u + threadIdx.x;
    if (e < (unsigned)N_EDGES) atomicAdd(&deg[ei[N_EDGES + e]], 1.0f);
}

__global__ void k_aggr(const float* __restrict__ h, const int* __restrict__ ei,
                       float* __restrict__ aggr) {
    unsigned i = blockIdx.x * 256u + threadIdx.x;   // exactly E*128 threads
    unsigned e = i >> 7, f = i & 127u;
    int s = ei[e];
    int d = ei[N_EDGES + e];
    atomicAdd(&aggr[(size_t)d * 128u + f], h[(size_t)s * 128u + f]);
}

// ---------------------------------------------------------------- SAGE layer: hout = (aggr/deg)@Wl + bl + h@Wr
__global__ void __launch_bounds__(256) k_sage(const float* __restrict__ aggr,
                                              const float* __restrict__ deg,
                                              const float* __restrict__ h,
                                              const float* __restrict__ Wl,
                                              const float* __restrict__ bl,
                                              const float* __restrict__ Wr,
                                              float* __restrict__ hout) {
    __shared__ bf16_t sWT[128 * 128];  // weight transposed [n][k] (re-staged: Wl then Wr)
    __shared__ bf16_t sAg[16 * 128];   // (aggr/deg) tile
    __shared__ bf16_t sH[16 * 128];    // h tile
    const unsigned tid = threadIdx.x;
    const unsigned mbase = blockIdx.x * 16u;

    for (unsigned idx = tid; idx < 16u * 128u; idx += 256u) {
        unsigned r = idx >> 7, k = idx & 127u;
        unsigned row = mbase + r;
        float inv = 1.0f / fmaxf(deg[row], 1.0f);
        sAg[idx] = (bf16_t)(aggr[(size_t)row * 128u + k] * inv);
        sH[idx]  = (bf16_t)h[(size_t)row * 128u + k];
    }
    for (unsigned idx = tid; idx < 128u * 128u; idx += 256u) {
        unsigned k = idx >> 7, n = idx & 127u;
        sWT[n * 128u + k] = (bf16_t)Wl[idx];
    }
    __syncthreads();

    const unsigned wave = tid >> 5, lane = tid & 31u;
    const unsigned half = lane >> 4, r = lane & 15u;
    const unsigned n = wave * 16u + r;

    v8f acc = {};
#pragma unroll
    for (unsigned kt = 0; kt < 4; ++kt) {
        const unsigned kb = kt * 32u;
        v16bf a, b;
#pragma unroll
        for (int e = 0; e < 8; ++e) {
            a[e]     = sAg[r * 128u + kb + half * 8u + e];
            a[e + 8] = sAg[r * 128u + kb + 16u + half * 8u + e];
        }
#pragma unroll
        for (int e = 0; e < 16; ++e)
            b[e] = sWT[n * 128u + kb + half * 8u + (e & 7) + (e >> 3) * 16u];
        acc = __builtin_amdgcn_wmma_f32_16x16x32_bf16(false, a, false, b,
                                                      (short)0, acc, false, false);
    }
    __syncthreads();  // all waves done reading Wl
    for (unsigned idx = tid; idx < 128u * 128u; idx += 256u) {
        unsigned k = idx >> 7, nn = idx & 127u;
        sWT[nn * 128u + k] = (bf16_t)Wr[idx];
    }
    __syncthreads();
#pragma unroll
    for (unsigned kt = 0; kt < 4; ++kt) {
        const unsigned kb = kt * 32u;
        v16bf a, b;
#pragma unroll
        for (int e = 0; e < 8; ++e) {
            a[e]     = sH[r * 128u + kb + half * 8u + e];
            a[e + 8] = sH[r * 128u + kb + 16u + half * 8u + e];
        }
#pragma unroll
        for (int e = 0; e < 16; ++e)
            b[e] = sWT[n * 128u + kb + half * 8u + (e & 7) + (e >> 3) * 16u];
        acc = __builtin_amdgcn_wmma_f32_16x16x32_bf16(false, a, false, b,
                                                      (short)0, acc, false, false);
    }
    const float bias = bl[n];
#pragma unroll
    for (int v = 0; v < 8; ++v) {
        unsigned row = mbase + half * 8u + (unsigned)v;
        hout[(size_t)row * 128u + n] = acc[v] + bias;
    }
}

// ---------------------------------------------------------------- BatchNorm (batch statistics over all nodes)
__global__ void k_bnstats(const float* __restrict__ h, float* __restrict__ bnsum,
                          float* __restrict__ bnsq) {
    unsigned f = threadIdx.x;                   // 128 threads, one feature each (coalesced rows)
    unsigned rbase = blockIdx.x * 200u;         // 250 blocks * 200 rows = 50000
    float s = 0.0f, q = 0.0f;
    for (unsigned r = 0; r < 200u; ++r) {
        float v = h[(size_t)(rbase + r) * 128u + f];
        s += v;
        q += v * v;
    }
    atomicAdd(&bnsum[f], s);
    atomicAdd(&bnsq[f], q);
}

__global__ void k_bnfin(const float* __restrict__ bnsum, const float* __restrict__ bnsq,
                        const float* __restrict__ gamma, const float* __restrict__ beta,
                        float* __restrict__ scale, float* __restrict__ shift) {
    unsigned f = threadIdx.x;  // 128 threads
    const float invN = 1.0f / (float)N_NODES;
    float mean = bnsum[f] * invN;
    float var  = bnsq[f] * invN - mean * mean;
    float sc = gamma[f] * rsqrtf(var + 1e-5f);
    scale[f] = sc;
    shift[f] = beta[f] - mean * sc;
}

__global__ void k_bnapply(float* __restrict__ h, const float* __restrict__ scale,
                          const float* __restrict__ shift) {
    unsigned i = blockIdx.x * 256u + threadIdx.x;  // exactly N*HID threads
    unsigned f = i & 127u;
    h[i] = fmaxf(h[i] * scale[f] + shift[f], 0.0f);
}

// ---------------------------------------------------------------- pooling + head
__global__ void k_cnt(const int* __restrict__ batch, float* __restrict__ gcnt) {
    unsigned nidx = blockIdx.x * 256u + threadIdx.x;
    if (nidx < (unsigned)N_NODES) atomicAdd(&gcnt[batch[nidx]], 1.0f);
}

__global__ void k_pool(const float* __restrict__ h, const int* __restrict__ batch,
                       float* __restrict__ g) {
    unsigned i = blockIdx.x * 256u + threadIdx.x;  // exactly N*HID threads
    unsigned nidx = i >> 7, f = i & 127u;
    int b = batch[nidx];
    atomicAdd(&g[(size_t)b * 128u + f], h[i]);
}

__global__ void __launch_bounds__(256) k_head(const float* __restrict__ g,
                                              const float* __restrict__ gcnt,
                                              const float* __restrict__ W1,
                                              const float* __restrict__ b1,
                                              const float* __restrict__ W2,
                                              const float* __restrict__ b2,
                                              float* __restrict__ out) {
    __shared__ float sT[N_GRAPHS * HID];
    unsigned tid = threadIdx.x;
    for (unsigned idx = tid; idx < (unsigned)(N_GRAPHS * HID); idx += 256u) {
        unsigned i = idx >> 7, j = idx & 127u;
        float inv = 1.0f / fmaxf(gcnt[i], 1.0f);
        float acc = b1[j];
        for (unsigned k = 0; k < 128u; ++k)
            acc += (g[i * 128u + k] * inv) * W1[k * 128u + j];
        sT[idx] = fmaxf(acc, 0.0f);
    }
    __syncthreads();
    for (unsigned idx = tid; idx < (unsigned)(N_GRAPHS * 2); idx += 256u) {
        unsigned i = idx >> 1, j = idx & 1u;
        float acc = b2[j];
        for (unsigned k = 0; k < 128u; ++k)
            acc += sT[i * 128u + k] * W2[k * 2u + j];
        out[i * 2u + j] = acc;
    }
}

// ---------------------------------------------------------------- launch
extern "C" void kernel_launch(void* const* d_in, const int* in_sizes, int n_in,
                              void* d_out, int out_size, void* d_ws, size_t ws_size,
                              hipStream_t stream) {
    const float* x     = (const float*)d_in[0];
    const int*   ei    = (const int*)d_in[1];
    const int*   batch = (const int*)d_in[2];
    const float* Wp    = (const float*)d_in[3];
    const float* bp    = (const float*)d_in[4];
    const float* Wl    = (const float*)d_in[5];
    const float* bl    = (const float*)d_in[6];
    const float* Wr    = (const float*)d_in[7];
    const float* gamma = (const float*)d_in[8];
    const float* beta  = (const float*)d_in[9];
    const float* W1    = (const float*)d_in[10];
    const float* b1    = (const float*)d_in[11];
    const float* W2    = (const float*)d_in[12];
    const float* b2    = (const float*)d_in[13];
    float* out = (float*)d_out;

    float* ws = (float*)d_ws;
    size_t off = 0;
    float* hA    = ws + off; off += (size_t)N_NODES * HID;
    float* hB    = ws + off; off += (size_t)N_NODES * HID;
    float* aggr  = ws + off; off += (size_t)N_NODES * HID;   // deg directly follows (zeroed together)
    float* deg   = ws + off; off += N_NODES;
    float* bnsum = ws + off; off += HID;                     // bnsq follows (zeroed together)
    float* bnsq  = ws + off; off += HID;
    float* bsc   = ws + off; off += HID;
    float* bsh   = ws + off; off += HID;
    float* g     = ws + off; off += (size_t)N_GRAPHS * HID;  // gcnt follows (zeroed together)
    float* gcnt  = ws + off; off += N_GRAPHS;

    const unsigned NH = (unsigned)N_NODES * HID;             // 6,400,000
    const unsigned MTILES = N_NODES / 16;                    // 3125

    k_proj<<<MTILES, 256, 0, stream>>>(x, Wp, bp, hA);

    for (int l = 0; l < N_LAYERS; ++l) {
        unsigned nz = NH + N_NODES;  // aggr + deg
        k_zero<<<(nz + 255u) / 256u, 256, 0, stream>>>(aggr, nz);
        k_deg<<<(N_EDGES + 255) / 256, 256, 0, stream>>>(ei, deg);
        k_aggr<<<((unsigned)N_EDGES * HID) / 256u, 256, 0, stream>>>(hA, ei, aggr);
        k_sage<<<MTILES, 256, 0, stream>>>(aggr, deg, hA,
                                           Wl + (size_t)l * HID * HID, bl + (size_t)l * HID,
                                           Wr + (size_t)l * HID * HID, hB);
        k_zero<<<1, 256, 0, stream>>>(bnsum, 2u * HID);
        k_bnstats<<<N_NODES / 200, 128, 0, stream>>>(hB, bnsum, bnsq);
        k_bnfin<<<1, 128, 0, stream>>>(bnsum, bnsq, gamma + (size_t)l * HID,
                                       beta + (size_t)l * HID, bsc, bsh);
        k_bnapply<<<NH / 256u, 256, 0, stream>>>(hB, bsc, bsh);
        float* t = hA; hA = hB; hB = t;
    }

    unsigned gz = (unsigned)N_GRAPHS * HID + N_GRAPHS;
    k_zero<<<(gz + 255u) / 256u, 256, 0, stream>>>(g, gz);
    k_cnt<<<(N_NODES + 255) / 256, 256, 0, stream>>>(batch, gcnt);
    k_pool<<<NH / 256u, 256, 0, stream>>>(hA, batch, g);
    k_head<<<1, 256, 0, stream>>>(g, gcnt, W1, b1, W2, b2, out);
}
